// Model_12747462935042
// MI455X (gfx1250) — matrix operand order repaired and checked by reading, not compiled, not run on hardware
//
#include <hip/hip_runtime.h>
#include <hip/hip_bf16.h>
#include <math.h>

// ---------------- problem constants (from reference) ----------------
#define BB     8
#define TTT    16
#define NN     1024
#define HH     128
#define HEADS  8
#define HD     16
#define SNP    4
#define NT     (SNP*NN)        // 4096
#define DEG    8
#define MROWS  (BB*NT)         // 32768 rows for the GAT GEMMs
#define EBLK   (NN*DEG + NN)   // 9216 edges per snapshot block (random + self)
#define SEQ_OUT 12
#define LRELU  0.2f
#define APAD   8               // LDS row padding (halves) to kill bank conflicts
#define ASTRIDE (HH + APAD)    // 136 halves = 272 B = 68 dwords per row

typedef __attribute__((ext_vector_type(16))) _Float16 v16h;
typedef __attribute__((ext_vector_type(8)))  _Float16 v8h;
typedef __attribute__((ext_vector_type(8)))  float    v8f;

// ---------------- generic f32 -> f16 convert ----------------
__global__ void k_f32_to_f16(const float* __restrict__ src, _Float16* __restrict__ dst, int n) {
    int i = blockIdx.x * blockDim.x + threadIdx.x;
    if (i < n) dst[i] = (_Float16)src[i];
}

// ---------------- weight transpose+convert: W[k][n] (f32) -> WhT[n][k] (f16) ----------------
__global__ void k_w_to_f16T(const float* __restrict__ src, _Float16* __restrict__ dst) {
    int i = blockIdx.x * blockDim.x + threadIdx.x;   // over 128*128
    if (i >= HH * HH) return;
    int k = i >> 7, n = i & (HH - 1);
    dst[n * HH + k] = (_Float16)src[i];
}

// ---------------- stage 0: x = inputs@W_in + b_in + spatial + temporal ----------------
__global__ void k_embed(const float* __restrict__ inp, const float* __restrict__ W_in,
                        const float* __restrict__ b_in, const float* __restrict__ sp,
                        const float* __restrict__ te, float* __restrict__ X) {
    long i = (long)blockIdx.x * blockDim.x + threadIdx.x;   // over B*T*N*H
    if (i >= (long)BB * TTT * NN * HH) return;
    int  h  = (int)(i & (HH - 1));
    long r  = i >> 7;            // b*T*N + t*N + n
    int  n  = (int)(r % NN);
    long bt = r / NN;
    int  t  = (int)(bt % TTT);
    const float* in2 = inp + r * 2;
    X[i] = in2[0] * W_in[h] + in2[1] * W_in[HH + h] + b_in[h]
         + sp[n * HH + h] + te[n * TTT + t];
}

// ---------------- per-iteration: gather frames (+last) into fs, emit f16 copy ----------------
__global__ void k_build_fs(const float* __restrict__ X, const float* __restrict__ last,
                           float* __restrict__ fsf, _Float16* __restrict__ fsh,
                           int left, int use_last) {
    long i = (long)blockIdx.x * blockDim.x + threadIdx.x;   // over B*SNP*N*H
    if (i >= (long)BB * SNP * NN * HH) return;
    int  h    = (int)(i & (HH - 1));
    long r    = i >> 7;          // b*SNP*N + slot*N + n
    int  n    = (int)(r % NN);
    long bs   = r / NN;
    int  slot = (int)(bs % SNP);
    int  b    = (int)(bs / SNP);
    float v;
    if (use_last && slot == 0) {
        v = last[((long)b * NN + n) * HH + h];
    } else {
        int frame = left + slot - use_last;
        v = X[(((long)(b * TTT + frame)) * NN + n) * HH + h];
    }
    fsf[i] = v;
    fsh[i] = (_Float16)v;
}

// ---------------- WMMA GEMM: Y[M x 128] = Xh[M x 128] @ W[128 x 128] (f16 in, f32 out) ----------------
// Block = 8 waves covering one 16-row stripe x all 8 column tiles.
// A tile (16x128, shared by all 8 waves) staged once in LDS (padded rows, conflict-free).
// B fragments are contiguous 32-byte loads from the pre-transposed WhT[n][k].
__global__ __launch_bounds__(256) void k_wmma_gemm(const _Float16* __restrict__ Xh,
                                                   const _Float16* __restrict__ WhT,
                                                   float* __restrict__ Y) {
    __shared__ _Float16 As[16 * ASTRIDE];
    int tid = threadIdx.x;
    int m0  = blockIdx.x * 16;
    {   // cooperative A-tile fill: 16 rows x 256B, 16B per thread
        int row = tid >> 4, chunk = tid & 15;
        const v8h* srcp = (const v8h*)(Xh + (size_t)(m0 + row) * HH) + chunk;
        *(v8h*)(As + row * ASTRIDE + chunk * 8) = *srcp;
    }
    __syncthreads();
    int wave = tid >> 5;          // 0..7 -> column tile
    int lane = tid & 31;
    int half = lane >> 4, l16 = lane & 15;
    int n0 = wave * 16;
    v8f c = {};
    #pragma unroll
    for (int k0 = 0; k0 < 128; k0 += 32) {
        // A fragment from LDS: row l16, K = k0 + 16*half + j (contiguous)
        const _Float16* arow = As + l16 * ASTRIDE + k0 + half * 16;
        v8h alo = *(const v8h*)(arow);
        v8h ahi = *(const v8h*)(arow + 8);
        v16h a;
        #pragma unroll
        for (int i = 0; i < 8; ++i) { a[i] = alo[i]; a[i + 8] = ahi[i]; }
        // B fragment: col n = n0+l16, K = k0 + 16*half + j  (contiguous in WhT)
        v16h bfr = *(const v16h*)(WhT + (size_t)(n0 + l16) * HH + k0 + half * 16);
        c = __builtin_amdgcn_wmma_f32_16x16x32_f16(false, a, false, bfr,
                                                   (short)0, c, false, false);
    }
    int row = m0 + half * 8, col = n0 + l16;
    #pragma unroll
    for (int r = 0; r < 8; ++r) Y[(size_t)(row + r) * HH + col] = c[r];
}

// ---------------- el/er attention logits per (node, head) ----------------
__global__ void k_eler(const float* __restrict__ feat, const float* __restrict__ al,
                       const float* __restrict__ ar, float* __restrict__ el,
                       float* __restrict__ er) {
    int i = blockIdx.x * blockDim.x + threadIdx.x;   // B*NT*HEADS
    if (i >= MROWS * HEADS) return;
    int head = i & 7;
    const float* f = feat + (size_t)(i >> 3) * HH + head * HD;
    float sl = 0.f, sr = 0.f;
    #pragma unroll
    for (int d = 0; d < HD; ++d) {
        sl += f[d] * al[head * HD + d];
        sr += f[d] * ar[head * HD + d];
    }
    el[i] = sl; er[i] = sr;
}

// ---------------- GAT edge-softmax aggregation, exploiting fixed edge structure ----------------
// dst node v=(t,n): 8 random in-edges at esrc[t*EBLK + n*DEG .. +7], self loop, temporal (t>0)
__global__ __launch_bounds__(128) void k_gat_agg(const float* __restrict__ feat,
        const float* __restrict__ el, const float* __restrict__ er,
        const int* __restrict__ esrc, const float* __restrict__ bias,
        float* __restrict__ out, int activate) {
    int bid = blockIdx.x;                 // b*NT + v
    int b   = bid >> 12;                  // / NT
    int v   = bid & (NT - 1);
    int t   = v >> 10;
    int n   = v & (NN - 1);
    int tid = threadIdx.x;
    int head = tid >> 4, d = tid & 15;

    int srcs[10];
    int cnt = 0;
    int base = t * EBLK + n * DEG;
    #pragma unroll
    for (int j = 0; j < DEG; ++j) srcs[cnt++] = esrc[base + j];
    srcs[cnt++] = v;                      // self loop
    if (t > 0) srcs[cnt++] = v - NN;      // temporal edge from previous snapshot

    float erv = er[(size_t)bid * HEADS + head];
    float ev[10];
    float mx = -1e30f;
    for (int j = 0; j < cnt; ++j) {
        float e = el[((size_t)b * NT + srcs[j]) * HEADS + head] + erv;
        e = (e >= 0.f) ? e : LRELU * e;
        ev[j] = e;
        mx = fmaxf(mx, e);
    }
    float ssum = 0.f, acc = 0.f;
    for (int j = 0; j < cnt; ++j) {
        float w = __expf(ev[j] - mx);
        ssum += w;
        acc  += w * feat[((size_t)b * NT + srcs[j]) * HH + head * HD + d];
    }
    float o = acc / ssum + bias[head * HD + d];
    if (activate) o = (o > 0.f) ? o : (__expf(o) - 1.f);   // ELU
    out[(size_t)bid * HH + tid] = o;
}

// ---------------- residual add + temporal attention pooling over SNP ----------------
__global__ __launch_bounds__(128) void k_pool(const float* __restrict__ g2,
        const float* __restrict__ fsf, const float* __restrict__ q,
        float* __restrict__ last) {
    __shared__ float red[128];
    __shared__ float sc[SNP];
    int bid = blockIdx.x;                 // b*N + n
    int b = bid >> 10, n = bid & (NN - 1);
    int tid = threadIdx.x;
    float hv[SNP];
    float qv = q[n * HH + tid];
    #pragma unroll
    for (int t = 0; t < SNP; ++t) {
        size_t idx = (((size_t)b * NT) + t * NN + n) * HH + tid;
        hv[t] = g2[idx] + fsf[idx];       // h = gat_out + residual
    }
    for (int t = 0; t < SNP; ++t) {       // scores[b,t,n] = <h, agg_q[n]>
        red[tid] = hv[t] * qv; __syncthreads();
        for (int s = 64; s > 0; s >>= 1) { if (tid < s) red[tid] += red[tid + s]; __syncthreads(); }
        if (tid == 0) sc[t] = red[0];
        __syncthreads();
    }
    float m = fmaxf(fmaxf(sc[0], sc[1]), fmaxf(sc[2], sc[3]));
    float w0 = __expf(sc[0] - m), w1 = __expf(sc[1] - m);
    float w2 = __expf(sc[2] - m), w3 = __expf(sc[3] - m);
    float inv = 1.f / (w0 + w1 + w2 + w3);
    float lv = (w0 * hv[0] + w1 * hv[1] + w2 * hv[2] + w3 * hv[3]) * inv;
    last[((size_t)b * NN + n) * HH + tid] = lv;
}

// ---------------- per-batch mean / inv-std over N*H ----------------
__global__ __launch_bounds__(256) void k_stats(const float* __restrict__ last,
                                               float* __restrict__ st) {
    __shared__ float rs[256], rq[256];
    int b = blockIdx.x, tid = threadIdx.x;
    const float* p = last + (size_t)b * NN * HH;
    float s = 0.f, q = 0.f;
    for (int i = tid; i < NN * HH; i += 256) { float v = p[i]; s += v; q += v * v; }
    rs[tid] = s; rq[tid] = q; __syncthreads();
    for (int k = 128; k > 0; k >>= 1) {
        if (tid < k) { rs[tid] += rs[tid + k]; rq[tid] += rq[tid + k]; }
        __syncthreads();
    }
    if (tid == 0) {
        const float cnt = (float)(NN * HH);
        float mu  = rs[0] / cnt;
        float var = rq[0] / cnt - mu * mu;
        st[b * 2]     = mu;
        st[b * 2 + 1] = rsqrtf(var + 1e-5f);
    }
}

__global__ void k_norm(float* __restrict__ last, const float* __restrict__ st) {
    long i = (long)blockIdx.x * blockDim.x + threadIdx.x;
    if (i >= (long)BB * NN * HH) return;
    int b = (int)(i >> 17);               // N*H = 131072 = 2^17
    last[i] = (last[i] - st[b * 2]) * st[b * 2 + 1];
}

// ---------------- output head: relu(last*w1+b1) reduced over H with w2 ----------------
__global__ __launch_bounds__(128) void k_head(const float* __restrict__ last,
        const float* __restrict__ w1, const float* __restrict__ b1,
        const float* __restrict__ w2, const float* __restrict__ b2,
        float* __restrict__ out) {
    __shared__ float red[128];
    int bid = blockIdx.x;                 // b*N + n
    int b = bid >> 10, n = bid & (NN - 1);
    int tid = threadIdx.x;                // h
    float lv  = last[((size_t)b * NN + n) * HH + tid];
    float w2h = w2[tid];
    float part[SEQ_OUT];
    #pragma unroll
    for (int s = 0; s < SEQ_OUT; ++s) {
        float o = lv * w1[s] + b1[s];
        o = (o > 0.f) ? o : 0.f;
        part[s] = o * w2h;
    }
    for (int s = 0; s < SEQ_OUT; ++s) {
        red[tid] = part[s]; __syncthreads();
        for (int k = 64; k > 0; k >>= 1) { if (tid < k) red[tid] += red[tid + k]; __syncthreads(); }
        if (tid == 0) out[((size_t)b * SEQ_OUT + s) * NN + n] = red[0] + b2[0];
        __syncthreads();
    }
}

// ---------------- host orchestration ----------------
extern "C" void kernel_launch(void* const* d_in, const int* in_sizes, int n_in,
                              void* d_out, int out_size, void* d_ws, size_t ws_size,
                              hipStream_t stream) {
    const float* inputs   = (const float*)d_in[0];
    const float* W_in     = (const float*)d_in[1];
    const float* b_in     = (const float*)d_in[2];
    const float* spatial  = (const float*)d_in[3];
    const float* temporal = (const float*)d_in[4];
    const float* gw0 = (const float*)d_in[5];
    const float* al0 = (const float*)d_in[6];
    const float* ar0 = (const float*)d_in[7];
    const float* gb0 = (const float*)d_in[8];
    const float* gw1 = (const float*)d_in[9];
    const float* al1 = (const float*)d_in[10];
    const float* ar1 = (const float*)d_in[11];
    const float* gb1 = (const float*)d_in[12];
    const float* agg_q  = (const float*)d_in[13];
    const float* w_out1 = (const float*)d_in[14];
    const float* b_out1 = (const float*)d_in[15];
    const float* w_out2 = (const float*)d_in[16];
    const float* b_out2 = (const float*)d_in[17];
    const int*   esrc   = (const int*)d_in[18];
    (void)in_sizes; (void)n_in; (void)ws_size;
    float* out = (float*)d_out; (void)out_size;

    // carve workspace (256-byte aligned slices)
    char* wsp = (char*)d_ws;
    auto carve = [&](size_t bytes) -> char* {
        char* p = wsp;
        wsp += (bytes + 255) & ~(size_t)255;
        return p;
    };
    float*    X     = (float*)   carve((size_t)BB * TTT * NN * HH * 4);   // 67 MB
    float*    fsf   = (float*)   carve((size_t)MROWS * HH * 4);           // 16.8 MB
    _Float16* fsh   = (_Float16*)carve((size_t)MROWS * HH * 2);
    float*    feat  = (float*)   carve((size_t)MROWS * HH * 4);
    _Float16* h16   = (_Float16*)carve((size_t)MROWS * HH * 2);
    float*    g1    = (float*)   carve((size_t)MROWS * HH * 4);
    float*    g2    = (float*)   carve((size_t)MROWS * HH * 4);
    float*    el    = (float*)   carve((size_t)MROWS * HEADS * 4);
    float*    er    = (float*)   carve((size_t)MROWS * HEADS * 4);
    float*    last  = (float*)   carve((size_t)BB * NN * HH * 4);
    float*    stats = (float*)   carve((size_t)BB * 2 * 4);
    _Float16* W0T   = (_Float16*)carve((size_t)HH * HH * 2);   // transposed f16 weights
    _Float16* W1T   = (_Float16*)carve((size_t)HH * HH * 2);

    // weights -> transposed f16 (once per launch; deterministic)
    k_w_to_f16T<<<(HH * HH + 255) / 256, 256, 0, stream>>>(gw0, W0T);
    k_w_to_f16T<<<(HH * HH + 255) / 256, 256, 0, stream>>>(gw1, W1T);

    // stage 0: embed
    {
        long n = (long)BB * TTT * NN * HH;
        k_embed<<<(unsigned)((n + 255) / 256), 256, 0, stream>>>(inputs, W_in, b_in,
                                                                 spatial, temporal, X);
    }

    const int lefts[5] = {0, 4, 7, 10, 13};
    const int gemm_blocks = MROWS / 16;                 // one 16-row stripe per block
    const long fs_elems = (long)MROWS * HH;

    for (int it = 0; it < 5; ++it) {
        k_build_fs<<<(unsigned)(fs_elems / 256), 256, 0, stream>>>(X, last, fsf, fsh,
                                                                   lefts[it], it > 0 ? 1 : 0);
        // ---- GAT layer 0 ----
        k_wmma_gemm<<<gemm_blocks, 256, 0, stream>>>(fsh, W0T, feat);
        k_eler<<<(MROWS * HEADS) / 256, 256, 0, stream>>>(feat, al0, ar0, el, er);
        k_gat_agg<<<MROWS, 128, 0, stream>>>(feat, el, er, esrc, gb0, g1, 1);
        k_f32_to_f16<<<(unsigned)(fs_elems / 256), 256, 0, stream>>>(g1, h16, (int)fs_elems);
        // ---- GAT layer 1 ----
        k_wmma_gemm<<<gemm_blocks, 256, 0, stream>>>(h16, W1T, feat);
        k_eler<<<(MROWS * HEADS) / 256, 256, 0, stream>>>(feat, al1, ar1, el, er);
        k_gat_agg<<<MROWS, 128, 0, stream>>>(feat, el, er, esrc, gb1, g2, 0);
        // ---- pooling + normalization ----
        k_pool<<<BB * NN, 128, 0, stream>>>(g2, fsf, agg_q, last);
        k_stats<<<BB, 256, 0, stream>>>(last, stats);
        k_norm<<<(BB * NN * HH) / 256, 256, 0, stream>>>(last, stats);
    }

    k_head<<<BB * NN, 128, 0, stream>>>(last, w_out1, b_out1, w_out2, b_out2, out);
}
